// MultiHeadAttention_47923245089150
// MI455X (gfx1250) — matrix operand order repaired and checked by reading, not compiled, not run on hardware
//
#include <hip/hip_runtime.h>
#include <hip/hip_bf16.h>

typedef __bf16 bf16;
typedef __attribute__((ext_vector_type(16))) __bf16 v16bf;
typedef __attribute__((ext_vector_type(8)))  __bf16 bf16x8;
typedef __attribute__((ext_vector_type(4)))  __bf16 bf16x4;
typedef __attribute__((ext_vector_type(8)))  float  v8f;
typedef __attribute__((ext_vector_type(4)))  float  f32x4;

#define BATCH   2
#define SEQ     4096
#define DMODEL  768
#define NHEADS  12
#define DK      64
#define NEGVAL  (-1e9f)

#define WMMA_BF16(a, b, c) \
  __builtin_amdgcn_wmma_f32_16x16x32_bf16(false, (a), false, (b), (short)0, (c), false, false)

static __device__ __forceinline__ v16bf cat8(bf16x8 lo, bf16x8 hi) {
  return __builtin_shufflevector(lo, hi, 0,1,2,3,4,5,6,7,8,9,10,11,12,13,14,15);
}

static __device__ __forceinline__ float hmax16(float v) {
  v = fmaxf(v, __shfl_xor(v, 1, 32));
  v = fmaxf(v, __shfl_xor(v, 2, 32));
  v = fmaxf(v, __shfl_xor(v, 4, 32));
  v = fmaxf(v, __shfl_xor(v, 8, 32));
  return v;
}
static __device__ __forceinline__ float hsum16(float v) {
  v += __shfl_xor(v, 1, 32);
  v += __shfl_xor(v, 2, 32);
  v += __shfl_xor(v, 4, 32);
  v += __shfl_xor(v, 8, 32);
  return v;
}

// ---------------------------------------------------------------------------
// f32 -> bf16 conversion (4-wide)
// ---------------------------------------------------------------------------
__global__ void cvt_f32_bf16(const float* __restrict__ in, bf16* __restrict__ out, int n4) {
  int i = blockIdx.x * blockDim.x + threadIdx.x;
  if (i >= n4) return;
  f32x4 v = ((const f32x4*)in)[i];
  bf16x4 o;
  o.x = (bf16)v.x; o.y = (bf16)v.y; o.z = (bf16)v.z; o.w = (bf16)v.w;
  ((bf16x4*)out)[i] = o;
}

// ---------------------------------------------------------------------------
// GEMM: C[m,n] = sum_k X[m,k] * W[n,k] + bias[n]     (torch Linear: x @ W.T + b)
// X: [8192 x 768] bf16 row-major.  W: [768 x 768] bf16 row-major (K contiguous).
// 4 waves/block, block computes 64x64 tile; wave -> 16 rows x 64 cols.
// W tile (64 cols x 32 k) staged in LDS via async global->LDS copies,
// double-buffered, shared by all 4 waves (removes 4x-redundant weight loads).
// The LDS destination VGPR is derived from the real &Wlds address (ptrtoint of
// the AS3->flat cast: low 32 bits == LDS offset), so the shared array escapes
// into the asm and the B-fragment ds_loads cannot be folded away.
// mode 0: Q  -> bf16 [B,H,S,DK], scaled by 1/8
// mode 1: K  -> bf16 [B,H,S,DK]
// mode 2: V  -> bf16 [B,H,DK,S]  (transposed for P*V B-fragments)
// mode 3: O  -> f32  [B,S,D]     (final output)
// ---------------------------------------------------------------------------
#define WROW 40  // padded row stride (elements): 80B -> conflict-free ds_load_b128

__global__ __launch_bounds__(128)
void gemm_wmma(const bf16* __restrict__ X, const bf16* __restrict__ W,
               const float* __restrict__ bias, bf16* __restrict__ outb,
               float* __restrict__ outf, int mode) {
  __shared__ __align__(16) bf16 Wlds[2][64][WROW];

  const int tid  = threadIdx.x;
  const int lane = tid & 31;
  const int wave = tid >> 5;
  const int nl   = lane & 15;
  const int h    = lane >> 4;

  const int rowBase = blockIdx.x * 64 + wave * 16;
  const int colBase = blockIdx.y * 64;
  const int m = rowBase + nl;                    // A-fragment row for this lane

  // Async stage of one 64x32 W tile: 128 threads, each copies 32B (2 x b128).
  const int cprow  = tid >> 1;                   // 0..63: W column within tile
  const int cphalf = tid & 1;                    // 0/1:   k-halves (16 elems each)
  const bf16* csrc0 = W + (size_t)(colBase + cprow) * DMODEL + 16 * cphalf;

#define STAGE_W(bufi, kb)                                                        \
  do {                                                                           \
    const bf16* _s = csrc0 + (kb);                                               \
    bf16* _dp = &Wlds[(bufi)][cprow][16 * cphalf];                               \
    unsigned _d = (unsigned)(size_t)_dp;  /* flat low 32 bits == LDS offset */   \
    asm volatile("global_load_async_to_lds_b128 %0, %1, off"                     \
                 :: "v"(_d), "v"(_s), "v"(_dp) : "memory");                      \
    asm volatile("global_load_async_to_lds_b128 %0, %1, off offset:16"           \
                 :: "v"(_d), "v"(_s), "v"(_dp) : "memory");                      \
  } while (0)

#define WAIT_ASYNC_BAR()                                                         \
  do {                                                                           \
    asm volatile("s_wait_asynccnt 0x0" ::: "memory");                            \
    __syncthreads();                                                             \
  } while (0)

  STAGE_W(0, 0);
  WAIT_ASYNC_BAR();

  v8f acc[4] = {};
  int buf = 0;
  for (int kb = 0; kb < DMODEL; kb += 32, buf ^= 1) {
    if (kb + 32 < DMODEL) STAGE_W(buf ^ 1, kb + 32);

    const bf16* arow = X + (size_t)m * DMODEL + kb;
    bf16x8 a_lo = *(const bf16x8*)(arow + 8 * h);
    bf16x8 a_hi = *(const bf16x8*)(arow + 16 + 8 * h);
    v16bf a = cat8(a_lo, a_hi);
#pragma unroll
    for (int t = 0; t < 4; ++t) {
      const bf16* wr = &Wlds[buf][16 * t + nl][8 * h];    // B col = 16t+nl, K 8h..8h+7, 16+8h..16+8h+7
      v16bf b = cat8(*(const bf16x8*)wr, *(const bf16x8*)(wr + 16));
      acc[t] = WMMA_BF16(a, b, acc[t]);
    }
    WAIT_ASYNC_BAR();   // next buffer fully written; all reads of it long done
  }

#pragma unroll
  for (int t = 0; t < 4; ++t) {
    const int gn = colBase + 16 * t + nl;
    const float bv = bias[gn];
    const int hh = gn >> 6;                      // head
    const int d  = gn & 63;                      // dim within head
#pragma unroll
    for (int r = 0; r < 8; ++r) {
      const int gm = rowBase + r + 8 * h;        // C row: VGPR r, half h
      const float v = acc[t][r] + bv;
      const int b = gm >> 12;                    // batch (S = 4096)
      const int s = gm & 4095;
      if (mode == 0) {
        outb[((size_t)((b * NHEADS + hh) * SEQ + s) << 6) + d] = (bf16)(v * 0.125f);
      } else if (mode == 1) {
        outb[((size_t)((b * NHEADS + hh) * SEQ + s) << 6) + d] = (bf16)v;
      } else if (mode == 2) {
        outb[((size_t)((b * NHEADS + hh) * DK + d) << 12) + s] = (bf16)v;
      } else {
        outf[(size_t)gm * DMODEL + gn] = v;
      }
    }
  }
#undef STAGE_W
#undef WAIT_ASYNC_BAR
}

// ---------------------------------------------------------------------------
// Flash attention. Q pre-scaled by 1/sqrt(DK).
// grid: (SEQ/64, BATCH*NHEADS), block 128 (4 waves). Wave owns 16 q rows.
// Q,K: [B,H,S,64] bf16.  Vt: [B,H,64,S] bf16.  O: [B,S,768] bf16.
// ---------------------------------------------------------------------------
__global__ __launch_bounds__(128)
void attn_wmma(const bf16* __restrict__ Q, const bf16* __restrict__ K,
               const bf16* __restrict__ Vt, const int* __restrict__ mask,
               bf16* __restrict__ O) {
  __shared__ __align__(16) bf16 Plds[4][16][40];   // per-wave 16x32 P tile, padded

  const int lane = threadIdx.x & 31;
  const int wave = threadIdx.x >> 5;
  const int nl   = lane & 15;
  const int h    = lane >> 4;

  const int bh = blockIdx.y;                 // b*NHEADS + head
  const int b  = bh / NHEADS;
  const int hd = bh - b * NHEADS;
  const int qbase = blockIdx.x * 64 + wave * 16;

  // Load Q A-fragments (16 rows x 64 dk) once, held in registers.
  const bf16* qrow = Q + (((size_t)bh * SEQ + qbase + nl) << 6);
  v16bf qa0 = *(const v16bf*)(qrow +      16 * h);
  v16bf qa1 = *(const v16bf*)(qrow + 32 + 16 * h);

  const bf16* Kp = K  + ((size_t)bh * SEQ << 6);
  const bf16* Vp = Vt + ((size_t)bh * DK << 12);
  const int* mp  = mask + b * SEQ;

  float m_run[8], l_run[8];
#pragma unroll
  for (int r = 0; r < 8; ++r) { m_run[r] = -1e30f; l_run[r] = 0.f; }
  v8f o[4] = {};

  const int kvlen  = qbase + 16;             // causal: keys 0 .. qbase+15
  const int ntiles = (kvlen + 31) >> 5;

  for (int kt = 0; kt < ntiles; ++kt) {
    const int kbase = kt << 5;

    // --- scores: two 16x16 tiles over 32 keys, K-dim = 64 (2 chained WMMAs) ---
    v8f s[2];
#pragma unroll
    for (int t = 0; t < 2; ++t) {
      const int key = kbase + 16 * t + nl;
      const int mv  = mp[key];
      const bf16* kr = Kp + ((size_t)key << 6);
      v16bf kb0 = *(const v16bf*)(kr + 16 * h);
      v16bf kb1 = *(const v16bf*)(kr + 32 + 16 * h);
      v8f z = {};
      z = WMMA_BF16(qa0, kb0, z);
      z = WMMA_BF16(qa1, kb1, z);
#pragma unroll
      for (int r = 0; r < 8; ++r) {
        const int qg = qbase + r + 8 * h;    // C row for VGPR r
        const bool bad = (key > qg) || (mv == 0);
        z[r] = bad ? NEGVAL : z[r];
      }
      s[t] = z;
    }

    // --- online softmax ---
    float newm[8], scale[8], psum[8];
#pragma unroll
    for (int r = 0; r < 8; ++r) {
      float tm = hmax16(fmaxf(s[0][r], s[1][r]));
      newm[r]  = fmaxf(m_run[r], tm);
      scale[r] = __expf(m_run[r] - newm[r]);
      m_run[r] = newm[r];
    }
#pragma unroll
    for (int r = 0; r < 8; ++r) {
      float p0 = __expf(s[0][r] - newm[r]);
      float p1 = __expf(s[1][r] - newm[r]);
      Plds[wave][r + 8 * h][nl]      = (bf16)p0;
      Plds[wave][r + 8 * h][16 + nl] = (bf16)p1;
      psum[r] = hsum16(p0 + p1);
      l_run[r] = l_run[r] * scale[r] + psum[r];
    }
    // rescale running output
#pragma unroll
    for (int t = 0; t < 4; ++t)
#pragma unroll
      for (int r = 0; r < 8; ++r) o[t][r] *= scale[r];

    // --- P (C-layout) -> A-layout via LDS, then P*V ---
    bf16x8 plo = *(const bf16x8*)&Plds[wave][nl][16 * h];
    bf16x8 phi = *(const bf16x8*)&Plds[wave][nl][16 * h + 8];
    v16bf pa = cat8(plo, phi);
#pragma unroll
    for (int t = 0; t < 4; ++t) {
      const bf16* vr = Vp + ((size_t)(16 * t + nl) << 12) + kbase + 16 * h;
      v16bf vb = *(const v16bf*)vr;
      o[t] = WMMA_BF16(pa, vb, o[t]);
    }
  }

  // --- epilogue: divide by l, store O in [B,S,768] bf16 for final projection ---
  float inv[8];
#pragma unroll
  for (int r = 0; r < 8; ++r) inv[r] = 1.f / l_run[r];
#pragma unroll
  for (int t = 0; t < 4; ++t) {
#pragma unroll
    for (int r = 0; r < 8; ++r) {
      const int qg = qbase + r + 8 * h;
      O[((size_t)b * SEQ + qg) * DMODEL + hd * DK + 16 * t + nl] = (bf16)(o[t][r] * inv[r]);
    }
  }
}

// ---------------------------------------------------------------------------
// Host launcher
// ---------------------------------------------------------------------------
extern "C" void kernel_launch(void* const* d_in, const int* in_sizes, int n_in,
                              void* d_out, int out_size, void* d_ws, size_t ws_size,
                              hipStream_t stream) {
  const float* query = (const float*)d_in[0];
  const float* key   = (const float*)d_in[1];
  const float* value = (const float*)d_in[2];
  const int*   mask  = (const int*)d_in[3];
  const float* Wq = (const float*)d_in[4];
  const float* bq = (const float*)d_in[5];
  const float* Wk = (const float*)d_in[6];
  const float* bk = (const float*)d_in[7];
  const float* Wv = (const float*)d_in[8];
  const float* bv = (const float*)d_in[9];
  const float* Wo = (const float*)d_in[10];
  const float* bo = (const float*)d_in[11];

  char* p = (char*)d_ws;
  const size_t szW = (size_t)DMODEL * DMODEL * sizeof(bf16);
  const size_t szX = (size_t)BATCH * SEQ * DMODEL * sizeof(bf16);
  bf16* Wq_b = (bf16*)p; p += szW;
  bf16* Wk_b = (bf16*)p; p += szW;
  bf16* Wv_b = (bf16*)p; p += szW;
  bf16* Wo_b = (bf16*)p; p += szW;
  bf16* Xq_b = (bf16*)p; p += szX;
  bf16* Xk_b = (bf16*)p; p += szX;
  bf16* Xv_b = (bf16*)p; p += szX;
  bf16* Qb   = (bf16*)p; p += szX;   // [B,H,S,DK]
  bf16* Kb   = (bf16*)p; p += szX;   // [B,H,S,DK]
  bf16* Vtb  = (bf16*)p; p += szX;   // [B,H,DK,S]
  bf16* Ob   = (bf16*)p; p += szX;   // [B,S,D]

  const int nW4 = DMODEL * DMODEL / 4;
  const int nX4 = BATCH * SEQ * DMODEL / 4;
  cvt_f32_bf16<<<(nW4 + 255) / 256, 256, 0, stream>>>(Wq, Wq_b, nW4);
  cvt_f32_bf16<<<(nW4 + 255) / 256, 256, 0, stream>>>(Wk, Wk_b, nW4);
  cvt_f32_bf16<<<(nW4 + 255) / 256, 256, 0, stream>>>(Wv, Wv_b, nW4);
  cvt_f32_bf16<<<(nW4 + 255) / 256, 256, 0, stream>>>(Wo, Wo_b, nW4);
  cvt_f32_bf16<<<(nX4 + 255) / 256, 256, 0, stream>>>(query, Xq_b, nX4);
  cvt_f32_bf16<<<(nX4 + 255) / 256, 256, 0, stream>>>(key,   Xk_b, nX4);
  cvt_f32_bf16<<<(nX4 + 255) / 256, 256, 0, stream>>>(value, Xv_b, nX4);

  dim3 gg(BATCH * SEQ / 64, DMODEL / 64);
  gemm_wmma<<<gg, 128, 0, stream>>>(Xq_b, Wq_b, bq, Qb,  nullptr, 0);
  gemm_wmma<<<gg, 128, 0, stream>>>(Xk_b, Wk_b, bk, Kb,  nullptr, 1);
  gemm_wmma<<<gg, 128, 0, stream>>>(Xv_b, Wv_b, bv, Vtb, nullptr, 2);

  attn_wmma<<<dim3(SEQ / 64, BATCH * NHEADS), 128, 0, stream>>>(Qb, Kb, Vtb, mask, Ob);

  gemm_wmma<<<gg, 128, 0, stream>>>(Ob, Wo_b, bo, nullptr, (float*)d_out, 3);

  (void)in_sizes; (void)n_in; (void)out_size; (void)ws_size;
}
